// DeepKalmanFilter_32452772888683
// MI455X (gfx1250) — matrix operand order repaired and checked by reading, not compiled, not run on hardware
//
#include <hip/hip_runtime.h>

typedef __attribute__((ext_vector_type(16))) _Float16 v16h;
typedef __attribute__((ext_vector_type(8)))  _Float16 h8;
typedef __attribute__((ext_vector_type(8)))  float    v8f;

#define T_LEN 512
#define BATCH 2048

// ---------------- fast activations (v_exp_f32 + v_rcp_f32) ----------------
__device__ __forceinline__ float fast_sig(float x){
  return __builtin_amdgcn_rcpf(1.0f + __builtin_amdgcn_exp2f(-1.44269504f * x));
}
__device__ __forceinline__ float fast_tanh(float x){
  return 2.0f * __builtin_amdgcn_rcpf(1.0f + __builtin_amdgcn_exp2f(-2.88539008f * x)) - 1.0f;
}

// LDS fence: order wave-private DS store->load pairs (hw + compiler barrier)
#define LDS_FENCE() asm volatile("s_wait_dscnt 0" ::: "memory")

// ---------------- WMMA fragment helpers (CDNA5 wave32 layouts) ----------------
// B fragment: B = W^T (K x N); W row-major [nvalid x ldw]; zero-padded.
// Lane L: n = L&15, kb = ((L>=16)?8:0), k = kb + e (e<8), 16 + kb + (e-8) (e>=8)  ->  b[e] = W[n][k]
__device__ __forceinline__ v16h load_B(const float* __restrict__ W, int ldw,
                                       int nvalid, int kvalid, int lane){
  const int n = lane & 15;
  const int kbase = (lane >> 4) << 3;
  v16h b;
#pragma unroll
  for (int e = 0; e < 16; ++e){
    const int k = (e < 8) ? kbase + e : 16 + kbase + (e - 8);
    const float v = (n < nvalid && k < kvalid) ? W[n * ldw + k] : 0.0f;
    b[e] = (_Float16)v;
  }
  return b;
}

// A fragment 16xK (K<=16, upper half of K=32 zero). Lane L: m = L&15,
// kb = ((L>=16)?8:0); a[e] = A[m][kb+e] for e<8, a[8..15] = 0.
__device__ __forceinline__ v16h a_from_k16(h8 lo, int lane, int kvalid){
  const int kb = (lane >> 4) << 3;
  v16h a;
#pragma unroll
  for (int e = 0; e < 8; ++e){
    a[e]     = (kb + e < kvalid) ? lo[e] : (_Float16)0.0f;
    a[e + 8] = (_Float16)0.0f;
  }
  return a;
}

// A fragment 16x32 from two 8-wide half-rows: k = kb+e (lo) and 16+kb+e (hi)
__device__ __forceinline__ v16h a_from_k32(h8 lo, h8 hi){
  v16h a;
#pragma unroll
  for (int e = 0; e < 8; ++e){ a[e] = lo[e]; a[e + 8] = hi[e]; }
  return a;
}

__device__ __forceinline__ v8f cinit(float b){
  v8f c;
#pragma unroll
  for (int v = 0; v < 8; ++v) c[v] = b;
  return c;
}

// D fragment (f32, 16x16: lane L holds col n=L&15, rows v + 8*(L>=16)) -> LDS f16 tile
__device__ __forceinline__ void store_frag(_Float16* buf, v8f c, int lane, int nvalid){
  const int n  = lane & 15;
  const int mh = (lane >> 4) << 3;
  if (n < nvalid){
#pragma unroll
    for (int v = 0; v < 8; ++v) buf[(mh + v) * 16 + n] = (_Float16)c[v];
  }
}
__device__ __forceinline__ void store_frag_tanh(_Float16* buf, v8f c, int lane){
  const int n  = lane & 15;
  const int mh = (lane >> 4) << 3;
#pragma unroll
  for (int v = 0; v < 8; ++v) buf[(mh + v) * 16 + n] = (_Float16)fast_tanh(c[v]);
}

// A half-row re-load from a 16x16 f16 LDS tile (ds_load_b128 per lane)
__device__ __forceinline__ h8 load_arow(const _Float16* buf, int lane){
  const int m  = lane & 15;
  const int kb = (lane >> 4) << 3;
  return *(const h8*)(buf + m * 16 + kb);
}

__device__ __forceinline__ v8f wmma16(v16h a, v16h b, v8f c){
  return __builtin_amdgcn_wmma_f32_16x16x32_f16(false, a, false, b, (short)0, c,
                                                false, false);
}

// =====================================================================
// Phase 1: backward LSTM (serial in t, parallel in batch: 128 waves).
// gates(16x64) = x_t * Wih^T + b + h * Whh^T  ->  4 WMMAs (K=16 padded)
// =====================================================================
__global__ void __launch_bounds__(128)
dkf_lstm_bwd(const float* __restrict__ x,
             const float* __restrict__ Wih,
             const float* __restrict__ Whh,
             const float* __restrict__ bih,
             const float* __restrict__ bhh,
             _Float16*    __restrict__ h_seq)
{
  const int lane = threadIdx.x & 31;
  const int wv   = threadIdx.x >> 5;
  const int mb   = blockIdx.x * 64 + wv * 16;
  const int n    = lane & 15;
  const int hi   = lane >> 4;
  const int mh   = hi << 3;

  __shared__ __align__(16) _Float16 sh[4][256];
  _Float16* buf = sh[wv];

  v16h Bg[4];
#pragma unroll
  for (int g = 0; g < 4; ++g) Bg[g] = load_B(Whh + (16 * g) * 16, 16, 16, 16, lane);

  float bias_g[4], wih_g[4];
#pragma unroll
  for (int g = 0; g < 4; ++g){
    bias_g[g] = bih[16 * g + n] + bhh[16 * g + n];
    wih_g[g]  = Wih[16 * g + n];          // X_DIM == 1
  }

  v8f cC; h8 hrow;
#pragma unroll
  for (int v = 0; v < 8; ++v) cC[v] = 0.0f;
#pragma unroll
  for (int e = 0; e < 8; ++e) hrow[e] = (_Float16)0.0f;

  for (int t = T_LEN - 1; t >= 0; --t){
    const v16h A = a_from_k16(hrow, lane, 16);

    float xv[8];
#pragma unroll
    for (int v = 0; v < 8; ++v) xv[v] = x[(size_t)t * BATCH + mb + mh + v];

    v8f gf[4];
#pragma unroll
    for (int g = 0; g < 4; ++g){
      v8f c;
#pragma unroll
      for (int v = 0; v < 8; ++v) c[v] = bias_g[g] + wih_g[g] * xv[v];
      gf[g] = wmma16(A, Bg[g], c);
    }

    v8f hC;
#pragma unroll
    for (int v = 0; v < 8; ++v){
      const float iv = fast_sig(gf[0][v]);
      const float fv = fast_sig(gf[1][v]);
      const float gv = fast_tanh(gf[2][v]);
      const float ov = fast_sig(gf[3][v]);
      const float cn = fv * cC[v] + iv * gv;
      cC[v] = cn;
      hC[v] = ov * fast_tanh(cn);
    }

    // h (D-layout) -> LDS f16 -> A half-row + stream to h_seq[t] (b128)
    store_frag(buf, hC, lane, 16);
    LDS_FENCE();
    hrow = load_arow(buf, lane);
    *(h8*)(h_seq + ((size_t)t * BATCH + mb + (lane & 15)) * 16 + hi * 8) = hrow;
  }
}

// =====================================================================
// Phase 2: forward DKF scan, RECURRENCE PATH ONLY (serial, 128 waves).
// combiner(K=32) -> comb2 -> enc1 -> enc2(x2) -> z : 5 WMMAs / step.
// Writes mu_z/logvar_z to out and z_seq (f16) for the parallel head.
// =====================================================================
__global__ void __launch_bounds__(128)
dkf_scan_fwd(const _Float16* __restrict__ h_seq,
             const float* __restrict__ eps_z,
             const float* __restrict__ comb_W1, const float* __restrict__ comb_b1,
             const float* __restrict__ comb_W2, const float* __restrict__ comb_b2,
             const float* __restrict__ enc_W1,  const float* __restrict__ enc_b1,
             const float* __restrict__ enc_W2,  const float* __restrict__ enc_b2,
             _Float16*    __restrict__ z_seq,
             float* __restrict__ out)
{
  const int lane = threadIdx.x & 31;
  const int wv   = threadIdx.x >> 5;
  const int mb   = blockIdx.x * 64 + wv * 16;
  const int n    = lane & 15;
  const int hi   = lane >> 4;
  const int mh   = hi << 3;

  __shared__ __align__(16) _Float16 shA[4][256];
  __shared__ __align__(16) _Float16 shZ[4][256];
  _Float16* bufA = shA[wv];
  _Float16* bufZ = shZ[wv];

  const v16h B_c1  = load_B(comb_W1,          32, 16, 32, lane); // single K=32 WMMA
  const v16h B_c2  = load_B(comb_W2,          16,  8, 16, lane);
  const v16h B_e1  = load_B(enc_W1,            8, 16,  8, lane);
  const v16h B_e2a = load_B(enc_W2,           16, 16, 16, lane);
  const v16h B_e2b = load_B(enc_W2 + 16 * 16, 16, 16, 16, lane);

  const float bb_c1  = comb_b1[n];
  const float bb_c2  = (n < 8) ? comb_b2[n] : 0.0f;
  const float bb_e1  = enc_b1[n];
  const float bb_e2a = enc_b2[n];
  const float bb_e2b = enc_b2[16 + n];

  h8 zrow;                    // z_prev A half-row (z_{-1} = 0)
#pragma unroll
  for (int e = 0; e < 8; ++e) zrow[e] = (_Float16)0.0f;

  for (int t = 0; t < T_LEN; ++t){
    const _Float16* hp = h_seq + ((size_t)t * BATCH + mb + (lane & 15)) * 16 + hi * 8;
    const h8 hrow = *(const h8*)hp;
    if (t + 1 < T_LEN)
      __builtin_prefetch(hp + (size_t)BATCH * 16, 0, 0);

    // combiner: hz = [h | z_prev] (K=32), then (16x16)x(16x8)
    v8f c = wmma16(a_from_k32(hrow, zrow), B_c1, cinit(bb_c1));
    store_frag_tanh(bufA, c, lane);              LDS_FENCE();
    const v16h A1 = a_from_k16(load_arow(bufA, lane), lane, 16);
    c = wmma16(A1, B_c2, cinit(bb_c2));          // g : cols 0..7 valid
    store_frag(bufA, c, lane, 8);                LDS_FENCE();
    const v16h A2 = a_from_k16(load_arow(bufA, lane), lane, 8);

    // encoder
    c = wmma16(A2, B_e1, cinit(bb_e1));
    store_frag_tanh(bufA, c, lane);              LDS_FENCE();
    const v16h A3 = a_from_k16(load_arow(bufA, lane), lane, 16);
    const v8f mu = wmma16(A3, B_e2a, cinit(bb_e2a));
    const v8f lv = wmma16(A3, B_e2b, cinit(bb_e2b));

    // reparameterize: z = mu + eps * exp(0.5*logvar)
    v8f z;
#pragma unroll
    for (int v = 0; v < 8; ++v){
      const float ev = eps_z[((size_t)t * BATCH + mb + mh + v) * 16 + n];
      z[v] = mu[v] + ev * __builtin_amdgcn_exp2f(0.72134752f * lv[v]);
    }

    // z -> LDS -> A half-row (recurrence) + stream z_seq[t] (b128, f16)
    store_frag(bufZ, z, lane, 16);               LDS_FENCE();
    zrow = load_arow(bufZ, lane);
    *(h8*)(z_seq + ((size_t)t * BATCH + mb + (lane & 15)) * 16 + hi * 8) = zrow;

    // out[t][b][66]: mu_z at +2, logvar_z at +18
#pragma unroll
    for (int v = 0; v < 8; ++v){
      const size_t row = ((size_t)t * BATCH + mb + mh + v) * 66;
      out[row +  2 + n] = mu[v];
      out[row + 18 + n] = lv[v];
    }
  }
}

// =====================================================================
// Phase 3: decoder + transition head — fully parallel over (t, b).
// 65536 independent waves, 5 WMMAs each; bandwidth-bound.
// =====================================================================
__global__ void __launch_bounds__(128)
dkf_head(const _Float16* __restrict__ z_seq,
         const float* __restrict__ dec_W1, const float* __restrict__ dec_b1,
         const float* __restrict__ dec_W2, const float* __restrict__ dec_b2,
         const float* __restrict__ tr_W1,  const float* __restrict__ tr_b1,
         const float* __restrict__ tr_W2,  const float* __restrict__ tr_b2,
         float* __restrict__ out)
{
  const int lane = threadIdx.x & 31;
  const int wv   = threadIdx.x >> 5;
  const int gid  = blockIdx.x * 4 + wv;       // tile id over (t, b/16)
  const int t    = gid >> 7;                  // BATCH/16 = 128 tiles per t
  const int mb   = (gid & 127) << 4;
  const int n    = lane & 15;
  const int hi   = lane >> 4;
  const int mh   = hi << 3;

  __shared__ __align__(16) _Float16 shD[4][256];
  __shared__ __align__(16) _Float16 shT[4][256];
  _Float16* bufD = shD[wv];
  _Float16* bufT = shT[wv];

  const v16h B_d1  = load_B(dec_W1,          16, 16, 16, lane);
  const v16h B_d2  = load_B(dec_W2,          16,  2, 16, lane);
  const v16h B_t1  = load_B(tr_W1,           16, 16, 16, lane);
  const v16h B_t2a = load_B(tr_W2,           16, 16, 16, lane);
  const v16h B_t2b = load_B(tr_W2 + 16 * 16, 16, 16, 16, lane);

  const float bb_d1  = dec_b1[n];
  const float bb_d2  = (n < 2) ? dec_b2[n] : 0.0f;
  const float bb_t1  = tr_b1[n];
  const float bb_t2a = tr_b2[n];
  const float bb_t2b = tr_b2[16 + n];

  // z_t and z_{t-1} A half-rows (b128 loads; z_{-1} = 0)
  const h8 zrow = *(const h8*)(z_seq + ((size_t)t * BATCH + mb + (lane & 15)) * 16 + hi * 8);
  h8 zprow;
  if (t > 0){
    zprow = *(const h8*)(z_seq + ((size_t)(t - 1) * BATCH + mb + (lane & 15)) * 16 + hi * 8);
  } else {
#pragma unroll
    for (int e = 0; e < 8; ++e) zprow[e] = (_Float16)0.0f;
  }

  // two independent chains: decoder(z) and transition(z_prev)
  v8f cd = wmma16(a_from_k16(zrow,  lane, 16), B_d1, cinit(bb_d1));
  v8f ct = wmma16(a_from_k16(zprow, lane, 16), B_t1, cinit(bb_t1));
  store_frag_tanh(bufD, cd, lane);
  store_frag_tanh(bufT, ct, lane);
  LDS_FENCE();
  const v16h Ad = a_from_k16(load_arow(bufD, lane), lane, 16);
  const v16h At = a_from_k16(load_arow(bufT, lane), lane, 16);

  const v8f d2  = wmma16(Ad, B_d2,  cinit(bb_d2));   // cols 0,1 = mu_x, logvar_x
  const v8f mtr = wmma16(At, B_t2a, cinit(bb_t2a));
  const v8f ltr = wmma16(At, B_t2b, cinit(bb_t2b));

  // out[t][b][66]: mu_x@0, logvar_x@1, mu_tr@34+n, logvar_tr@50+n
#pragma unroll
  for (int v = 0; v < 8; ++v){
    const size_t row = ((size_t)t * BATCH + mb + mh + v) * 66;
    out[row + 34 + n] = mtr[v];
    out[row + 50 + n] = ltr[v];
    if (n < 2) out[row + n] = d2[v];
  }
}

extern "C" void kernel_launch(void* const* d_in, const int* in_sizes, int n_in,
                              void* d_out, int out_size, void* d_ws, size_t ws_size,
                              hipStream_t stream){
  (void)in_sizes; (void)n_in; (void)out_size; (void)ws_size;
  const float* x        = (const float*)d_in[0];
  const float* eps_z    = (const float*)d_in[1];
  const float* lstm_Wih = (const float*)d_in[2];
  const float* lstm_Whh = (const float*)d_in[3];
  const float* lstm_bih = (const float*)d_in[4];
  const float* lstm_bhh = (const float*)d_in[5];
  const float* comb_W1  = (const float*)d_in[6];
  const float* comb_b1  = (const float*)d_in[7];
  const float* comb_W2  = (const float*)d_in[8];
  const float* comb_b2  = (const float*)d_in[9];
  const float* enc_W1   = (const float*)d_in[10];
  const float* enc_b1   = (const float*)d_in[11];
  const float* enc_W2   = (const float*)d_in[12];
  const float* enc_b2   = (const float*)d_in[13];
  const float* dec_W1   = (const float*)d_in[14];
  const float* dec_b1   = (const float*)d_in[15];
  const float* dec_W2   = (const float*)d_in[16];
  const float* dec_b2   = (const float*)d_in[17];
  const float* tr_W1    = (const float*)d_in[18];
  const float* tr_b1    = (const float*)d_in[19];
  const float* tr_W2    = (const float*)d_in[20];
  const float* tr_b2    = (const float*)d_in[21];

  _Float16* h_seq = (_Float16*)d_ws;                                 // 32 MB
  _Float16* z_seq = (_Float16*)d_ws + (size_t)T_LEN * BATCH * 16;    // 32 MB
  float*    out   = (float*)d_out;

  dim3 block(128);
  dim3 gridScan(BATCH / 64);                    // 128 waves, serial phases
  dim3 gridHead(T_LEN * BATCH / 64);            // 65536 waves, parallel head

  dkf_lstm_bwd<<<gridScan, block, 0, stream>>>(x, lstm_Wih, lstm_Whh,
                                               lstm_bih, lstm_bhh, h_seq);
  dkf_scan_fwd<<<gridScan, block, 0, stream>>>(h_seq, eps_z,
                                               comb_W1, comb_b1, comb_W2, comb_b2,
                                               enc_W1, enc_b1, enc_W2, enc_b2,
                                               z_seq, out);
  dkf_head<<<gridHead, block, 0, stream>>>(z_seq,
                                           dec_W1, dec_b1, dec_W2, dec_b2,
                                           tr_W1, tr_b1, tr_W2, tr_b2, out);
}
